// KolmogorovArnoldLayer_38603166056799
// MI455X (gfx1250) — matrix operand order repaired and checked by reading, not compiled, not run on hardware
//
#include <hip/hip_runtime.h>
#include <hip/hip_bf16.h>

// ---------------------------------------------------------------------------
// KAN layer as one f16 WMMA GEMM:
//   out[B,O] = Aext[B, I*64] @ Wext[I*64, O]
//   Aext slot 0  = silu(x[b,i]);  slots 1..60 = cubic B-spline basis (<=4 nz)
//   Wext slot 0  = wb[i,o];       slots 1..60 = ws[i,o]*cps[o,k]
// B=2048, I=256, O=256 -> GEMM 2048 x 16384 x 256 (17.2 GFLOP f16 WMMA).
// Aext (64MB f16) + Wext (8MB f16) live in d_ws; Aext fits in the 192MB L2.
// GEMM pipeline: double-buffered LDS tiles filled with CDNA5 async
// global->LDS copies (ASYNCcnt) overlapping the WMMA stream.
// ---------------------------------------------------------------------------

typedef __attribute__((ext_vector_type(16))) _Float16 v16h;
typedef __attribute__((ext_vector_type(8)))  _Float16 v8h;
typedef __attribute__((ext_vector_type(8)))  float    v8f;

#define B_N   2048
#define I_N   256
#define O_N   256
#define KB    60              // number of basis functions
#define KSLOT 64              // 1 (silu) + 60 (basis) + 3 pad
#define KEXT  (I_N * KSLOT)   // 16384

// GEMM tiling
#define TM 128
#define TN 128
#define TK 32
#define KSPLIT 16
#define KCHUNK (KEXT / KSPLIT)   // 1024 -> 32 K-steps per block
#define LDA_P 40    // LDS pitch (halfs) for A tile rows: 80B -> conflict-free b128 reads
#define LDB_P 136   // LDS pitch (halfs) for B tile rows: 272B -> conflict-free b128 reads

// CDNA5 async global->LDS copies (guarded so host pass / older toolchains
// fall back to the proven load+ds_store path).
#if __has_builtin(__builtin_amdgcn_global_load_async_to_lds_b128) && \
    __has_builtin(__builtin_amdgcn_s_wait_asynccnt)
#define USE_ASYNC_LDS 1
#else
#define USE_ASYNC_LDS 0
#endif

#if USE_ASYNC_LDS
typedef int v4i __attribute__((vector_size(16)));
typedef __attribute__((address_space(1))) v4i* g4ptr;   // global (AS1)
typedef __attribute__((address_space(3))) v4i* l4ptr;   // LDS    (AS3)
#endif

__device__ __forceinline__ void copy16_g2l(const _Float16* g, _Float16* l) {
#if USE_ASYNC_LDS
    __builtin_amdgcn_global_load_async_to_lds_b128((g4ptr)g, (l4ptr)l, 0, 0);
#else
    *(v8h*)l = *(const v8h*)g;
#endif
}

__device__ __forceinline__ void wait_async_copies() {
#if USE_ASYNC_LDS
    __builtin_amdgcn_s_wait_asynccnt(0);
#endif
}

// ---------------------------------------------------------------------------
__global__ void zero_out_kernel(float* __restrict__ out, int n) {
    int t = blockIdx.x * blockDim.x + threadIdx.x;
    if (t < n) out[t] = 0.0f;
}

// Build Wext [KEXT, O] f16 row-major (k-major so B fragments load contiguously).
// One thread -> one contiguous v8h (16B) store of W[kext][o0..o0+7].
__global__ void prep_w_kernel(const float* __restrict__ wb,
                              const float* __restrict__ ws,
                              const float* __restrict__ cps,
                              _Float16* __restrict__ W) {
    int t = blockIdx.x * blockDim.x + threadIdx.x;  // over KEXT * (O_N/8)
    if (t >= KEXT * (O_N / 8)) return;
    int oc   = t & (O_N / 8 - 1);    // 0..31
    int o0   = oc * 8;
    int kext = t >> 5;               // 0..16383
    int i    = kext >> 6;
    int s    = kext & 63;

    alignas(16) _Float16 vals[8];
    if (s == 0) {
#pragma unroll
        for (int j = 0; j < 8; ++j)
            vals[j] = (_Float16)wb[i * O_N + o0 + j];
    } else if (s <= KB) {
        int k = s - 1;
#pragma unroll
        for (int j = 0; j < 8; ++j)
            vals[j] = (_Float16)(ws[i * O_N + o0 + j] * cps[(o0 + j) * 64 + k]);
    } else {
#pragma unroll
        for (int j = 0; j < 8; ++j) vals[j] = (_Float16)0.0f;
    }
    *(v8h*)(W + (size_t)kext * O_N + o0) = *(const v8h*)vals;
}

// Build Aext [B, KEXT] f16 row-major. Uniform (non-clamped) knots => every
// basis fn is the shifted cardinal cubic B-spline; closed form per span.
__global__ void prep_a_kernel(const float* __restrict__ X,
                              const float* __restrict__ knots,
                              _Float16* __restrict__ A) {
    int t = blockIdx.x * blockDim.x + threadIdx.x;  // over B*I
    if (t >= B_N * I_N) return;
    int i = t % I_N;
    int b = t / I_N;

    float x   = X[t];
    float k0  = knots[0];
    float h   = knots[1] - knots[0];
    float u   = (x - k0) / h;                 // span coordinate
    float sil = x / (1.0f + __expf(-x));      // silu(x)

    float v0 = 0.f, v1 = 0.f, v2 = 0.f, v3 = 0.f;
    int   j  = -1000;
    if (u >= 0.0f && u < 63.0f) {             // x in [knots[0], knots[63])
        j = (int)u;                           // floor (u >= 0)
        float tt  = u - (float)j;
        float omt = 1.0f - tt;
        float t2  = tt * tt, t3 = t2 * tt;
        const float s = 1.0f / 6.0f;
        v0 = omt * omt * omt * s;                             // m = j-3
        v1 = (3.0f * t3 - 6.0f * t2 + 4.0f) * s;              // m = j-2
        v2 = (-3.0f * t3 + 3.0f * t2 + 3.0f * tt + 1.0f) * s; // m = j-1
        v3 = t3 * s;                                          // m = j
    }

    alignas(16) _Float16 row[KSLOT];
    row[0] = (_Float16)sil;
#pragma unroll
    for (int ss = 1; ss <= KB; ++ss) {
        int m = ss - 1;
        int d = j - m;            // 0..3 picks v3..v0
        float v = 0.0f;
        v = (d == 0) ? v3 : v;
        v = (d == 1) ? v2 : v;
        v = (d == 2) ? v1 : v;
        v = (d == 3) ? v0 : v;
        row[ss] = (_Float16)v;
    }
#pragma unroll
    for (int ss = KB + 1; ss < KSLOT; ++ss) row[ss] = (_Float16)0.0f;

    _Float16* dst = A + (size_t)b * KEXT + i * KSLOT;   // 128B aligned
    const v8h* src = (const v8h*)row;
#pragma unroll
    for (int c = 0; c < 8; ++c) *((v8h*)dst + c) = src[c];
}

// ---------------------------------------------------------------------------
// GEMM: out += Aext @ Wext  (split-K partials via global f32 atomics).
// Double-buffered LDS; async copies of tile k+1 overlap WMMAs of tile k.
__global__ __launch_bounds__(256) void kan_gemm_kernel(const _Float16* __restrict__ A,
                                                       const _Float16* __restrict__ W,
                                                       float* __restrict__ out) {
    __shared__ alignas(16) _Float16 sA[2][TM * LDA_P];  // 2 x 10240 B
    __shared__ alignas(16) _Float16 sB[2][TK * LDB_P];  // 2 x  8704 B

    const int t    = threadIdx.x;
    const int lane = t & 31;
    const int wave = t >> 5;       // 0..7
    const int wm   = wave & 3;     // wave M offset: wm*32
    const int wn   = wave >> 2;    // wave N offset: wn*64
    const int m0   = blockIdx.x * TM;
    const int n0   = blockIdx.y * TN;
    const int kOff = blockIdx.z * KCHUNK;

    const int lrow = lane & 15;
    const int lhiA = (lane >> 4) << 3;   // A: hi half-wave -> K+8
    const int lhiB = (lane >> 4) << 4;   // B: hi half-wave -> K row +16

    // staging coordinates (per thread: 2 x 16B for A, 2 x 16B for B)
    const int aCol = (t & 3) * 8;
    const int aRow = t >> 2;             // 0..63 (+64 second half)
    const int bCol = (t & 15) * 8;
    const int bRow = t >> 4;             // 0..15 (+16 second half)

    v8f acc[2][4] = {};

    // prologue: fill buffer 0
    {
        const int kg = kOff;
#pragma unroll
        for (int hh = 0; hh < 2; ++hh) {
            const int r = aRow + hh * 64;
            copy16_g2l(A + (size_t)(m0 + r) * KEXT + kg + aCol,
                       &sA[0][r * LDA_P + aCol]);
        }
#pragma unroll
        for (int hh = 0; hh < 2; ++hh) {
            const int r = bRow + hh * 16;
            copy16_g2l(W + (size_t)(kg + r) * O_N + n0 + bCol,
                       &sB[0][r * LDB_P + bCol]);
        }
        wait_async_copies();
        __syncthreads();
    }

    int cur = 0;
    for (int ks = 0; ks < KCHUNK; ks += TK) {
        // ---- issue async copies for the next tile into the other buffer
        if (ks + TK < KCHUNK) {
            const int kg  = kOff + ks + TK;
            const int nxt = cur ^ 1;
#pragma unroll
            for (int hh = 0; hh < 2; ++hh) {
                const int r = aRow + hh * 64;
                copy16_g2l(A + (size_t)(m0 + r) * KEXT + kg + aCol,
                           &sA[nxt][r * LDA_P + aCol]);
            }
#pragma unroll
            for (int hh = 0; hh < 2; ++hh) {
                const int r = bRow + hh * 16;
                copy16_g2l(W + (size_t)(kg + r) * O_N + n0 + bCol,
                           &sB[nxt][r * LDB_P + bCol]);
            }
        }

        // ---- compute on current buffer: load fragments + 8 WMMAs
        v16h afrag[2], bfrag[4];
#pragma unroll
        for (int mi = 0; mi < 2; ++mi) {
            const int base = (wm * 32 + mi * 16 + lrow) * LDA_P + lhiA;
            v8h lo = *(const v8h*)(&sA[cur][base]);        // K 0..7  (+lhiA)
            v8h hi = *(const v8h*)(&sA[cur][base + 16]);   // K 16..23(+lhiA)
            afrag[mi] = __builtin_shufflevector(lo, hi, 0,1,2,3,4,5,6,7,
                                                        8,9,10,11,12,13,14,15);
        }
#pragma unroll
        for (int ni = 0; ni < 4; ++ni) {
            const int base = wn * 64 + ni * 16 + lrow;
#pragma unroll
            for (int s = 0; s < 16; ++s) {
                const int k = (s < 8) ? (lhiA + s) : (16 + lhiA + s - 8);
                bfrag[ni][s] = sB[cur][k * LDB_P + base];
            }
        }
#pragma unroll
        for (int mi = 0; mi < 2; ++mi)
#pragma unroll
            for (int ni = 0; ni < 4; ++ni)
                acc[mi][ni] = __builtin_amdgcn_wmma_f32_16x16x32_f16(
                    false, afrag[mi], false, bfrag[ni],
                    (short)0, acc[mi][ni], false, false);

        // ---- next tile fully in LDS before anyone reads it
        wait_async_copies();
        __syncthreads();
        cur ^= 1;
    }

    // ---- epilogue: split-K accumulate (C/D: lane 0-15 N=lane, vgpr r -> M=r / 8+r)
#pragma unroll
    for (int mi = 0; mi < 2; ++mi)
#pragma unroll
        for (int ni = 0; ni < 4; ++ni)
#pragma unroll
            for (int r = 0; r < 8; ++r) {
                const int row = m0 + wm * 32 + mi * 16 + ((lane >> 4) << 3) + r;
                const int col = n0 + wn * 64 + ni * 16 + (lane & 15);
                __hip_atomic_fetch_add(&out[row * O_N + col], acc[mi][ni][r],
                                       __ATOMIC_RELAXED, __HIP_MEMORY_SCOPE_AGENT);
            }
}

// ---------------------------------------------------------------------------
extern "C" void kernel_launch(void* const* d_in, const int* in_sizes, int n_in,
                              void* d_out, int out_size, void* d_ws, size_t ws_size,
                              hipStream_t stream) {
    const float* x     = (const float*)d_in[0];  // [2048,256]
    const float* wb    = (const float*)d_in[1];  // [256,256]
    const float* ws    = (const float*)d_in[2];  // [256,256]
    const float* cps   = (const float*)d_in[3];  // [256,64]
    const float* knots = (const float*)d_in[4];  // [64]
    // d_in[5] = degree (fixed 3)
    float* out = (float*)d_out;                  // [2048,256] f32

    _Float16* W = (_Float16*)d_ws;                                        // 8 MB
    _Float16* A = (_Float16*)((char*)d_ws +
                              (size_t)KEXT * O_N * sizeof(_Float16));     // 64 MB

    zero_out_kernel<<<(B_N * O_N + 255) / 256, 256, 0, stream>>>(out, B_N * O_N);
    prep_w_kernel<<<(KEXT * (O_N / 8) + 255) / 256, 256, 0, stream>>>(wb, ws, cps, W);
    prep_a_kernel<<<(B_N * I_N + 127) / 128, 128, 0, stream>>>(x, knots, A);

    dim3 grid(B_N / TM, O_N / TN, KSPLIT);   // 16 x 2 x 16 = 512 blocks
    kan_gemm_kernel<<<grid, 256, 0, stream>>>(A, W, out);
}
